// SelfAttention_61332132987328
// MI455X (gfx1250) — matrix operand order repaired and checked by reading, not compiled, not run on hardware
//
#include <hip/hip_runtime.h>
#include <hip/hip_bf16.h>

typedef __bf16 bf16_t;
typedef __attribute__((ext_vector_type(16))) __bf16 v16bf;
typedef __attribute__((ext_vector_type(8)))  __bf16 v8bf;
typedef __attribute__((ext_vector_type(8)))  float   v8f;

#define NB    2
#define NH    12
#define SEQ   2048
#define SREF  512
#define SK    2560          // SEQ + SREF
#define DH    64
#define CDIM  768
#define NEGB  (-1e30f)

#define WMMA_BF16(a, b, c) \
  __builtin_amdgcn_wmma_f32_16x16x32_bf16(false, (a), false, (b), (short)0, (c), false, false)

static __device__ __forceinline__ v8bf ld8(const bf16_t* p) { return *(const v8bf*)p; }

static __device__ __forceinline__ v16bf cat16(v8bf lo, v8bf hi) {
  v16bf r;
#pragma unroll
  for (int i = 0; i < 8; ++i) { r[i] = lo[i]; r[i + 8] = hi[i]; }
  return r;
}

static __device__ __forceinline__ v8f vzero8() {
  v8f r;
#pragma unroll
  for (int i = 0; i < 8; ++i) r[i] = 0.0f;
  return r;
}

static __device__ __forceinline__ unsigned pack2bf(float a, float b) {
  union { __bf16 h[2]; unsigned u; } z;
  z.h[0] = (__bf16)a; z.h[1] = (__bf16)b;
  return z.u;
}

// ---------------------------------------------------------------- convert ---
__global__ void cvt_f32_bf16(const float* __restrict__ s, bf16_t* __restrict__ d, int n) {
  int i = blockIdx.x * blockDim.x + threadIdx.x;
  if (i < n) d[i] = (bf16_t)s[i];
}

// ------------------------------------------------------------------- GEMM ---
// Y[m,n] = (sum_c X[m,c] * W[n,c] + bias[n]) * scale
// One wave computes a 16(m) x 64(n) tile; the k-loop is ping-pong
// software-pipelined (2 fragment sets, loads write straight into the set
// that was just consumed -> no register rotation copies).
// Requires Kd/32 even (Kd = 768 here).
// mode 0: fp32 row-major MxN  (final projection)
// mode 1: bf16 -> (B,NH,Lk,DH) head layout, rows offset by rowOff   (Q / K / RK)
// mode 2: bf16 -> (B,NH,DH,Lk) transposed head layout               (V / RV)
// linSh: log2 of input sequence length (rows per batch in X)
__global__ __launch_bounds__(128)
void gemm16x64(const bf16_t* __restrict__ X, const bf16_t* __restrict__ W,
               const float* __restrict__ bias, int M, int Kd, int N,
               float scale, int mode, int linSh, int Lk, int rowOff,
               void* __restrict__ outp) {
  const int lane = threadIdx.x & 31;
  const int wave = threadIdx.x >> 5;
  const int tiles_n = N >> 6;
  const int tile = blockIdx.x * 4 + wave;
  const int m0 = (tile / tiles_n) << 4;
  const int n0 = (tile % tiles_n) << 6;
  const int ml = lane & 15;
  const bool hi = lane >= 16;

  v8f acc[4];
#pragma unroll
  for (int i = 0; i < 4; ++i) acc[i] = vzero8();

  // A: 16x32 bf16, lane(<16)=row m0+ml, K chunks {0..7,16..23} (+8 for hi half)
  const bf16_t* xrow = X + (size_t)(m0 + ml) * Kd + (hi ? 8 : 0);
  const bf16_t* wp[4];
#pragma unroll
  for (int nt = 0; nt < 4; ++nt)
    wp[nt] = W + (size_t)(n0 + nt * 16 + ml) * Kd + (hi ? 8 : 0);

  const int nk = Kd >> 5;     // even
  v16bf a0 = cat16(ld8(xrow), ld8(xrow + 16));
  v16bf b0[4];
#pragma unroll
  for (int nt = 0; nt < 4; ++nt) b0[nt] = cat16(ld8(wp[nt]), ld8(wp[nt] + 16));

  int i = 0;
  for (; i + 2 < nk; i += 2) {
    const int k1 = (i + 1) << 5;
    // loads for chunk i+1 -> set 1 (overlap with set-0 WMMAs)
    v16bf a1 = cat16(ld8(xrow + k1), ld8(xrow + k1 + 16));
    v16bf b1[4];
#pragma unroll
    for (int nt = 0; nt < 4; ++nt)
      b1[nt] = cat16(ld8(wp[nt] + k1), ld8(wp[nt] + k1 + 16));
#pragma unroll
    for (int nt = 0; nt < 4; ++nt) acc[nt] = WMMA_BF16(a0, b0[nt], acc[nt]);
    // loads for chunk i+2 -> set 0 (just consumed; overlap with set-1 WMMAs)
    const int k2 = (i + 2) << 5;
    a0 = cat16(ld8(xrow + k2), ld8(xrow + k2 + 16));
#pragma unroll
    for (int nt = 0; nt < 4; ++nt)
      b0[nt] = cat16(ld8(wp[nt] + k2), ld8(wp[nt] + k2 + 16));
#pragma unroll
    for (int nt = 0; nt < 4; ++nt) acc[nt] = WMMA_BF16(a1, b1[nt], acc[nt]);
  }
  {  // last pair: chunk nk-2 (already in set 0) and chunk nk-1
    const int k1 = (i + 1) << 5;
    v16bf a1 = cat16(ld8(xrow + k1), ld8(xrow + k1 + 16));
    v16bf b1[4];
#pragma unroll
    for (int nt = 0; nt < 4; ++nt)
      b1[nt] = cat16(ld8(wp[nt] + k1), ld8(wp[nt] + k1 + 16));
#pragma unroll
    for (int nt = 0; nt < 4; ++nt) acc[nt] = WMMA_BF16(a0, b0[nt], acc[nt]);
#pragma unroll
    for (int nt = 0; nt < 4; ++nt) acc[nt] = WMMA_BF16(a1, b1[nt], acc[nt]);
  }

  const int rbase = hi ? 8 : 0;
  const int linMask = (1 << linSh) - 1;
#pragma unroll
  for (int nt = 0; nt < 4; ++nt) {
    const int n = n0 + nt * 16 + ml;
    const float bv = bias[n];
    if (mode == 0) {
      float* o = (float*)outp;
#pragma unroll
      for (int r = 0; r < 8; ++r)
        o[(size_t)(m0 + rbase + r) * N + n] = (acc[nt][r] + bv) * scale;
    } else if (mode == 1) {
      bf16_t* o = (bf16_t*)outp;
      const int hh = n >> 6, d = n & 63;
#pragma unroll
      for (int r = 0; r < 8; ++r) {
        const int m = m0 + rbase + r;
        const int bb = m >> linSh, t = m & linMask;
        o[(((size_t)bb * NH + hh) * Lk + (t + rowOff)) * DH + d] =
            (bf16_t)((acc[nt][r] + bv) * scale);
      }
    } else {
      bf16_t* o = (bf16_t*)outp;
      const int hh = n >> 6, d = n & 63;
      const int mb = m0 + rbase;                 // 8 consecutive rows, same batch
      const int bb = mb >> linSh, t = mb & linMask;
      v8bf wv;
#pragma unroll
      for (int r = 0; r < 8; ++r) wv[r] = (bf16_t)((acc[nt][r] + bv) * scale);
      *(v8bf*)(o + (((size_t)bb * NH + hh) * DH + d) * Lk + (t + rowOff)) = wv;
    }
  }
}

// -------------------------------------------------------- flash attention ---
// Computes S^T = K @ Q^T so the C-layout puts queries in lanes:
// softmax stats are per-lane (one shfl_xor(16) merges the key split),
// and O^T = V^T @ P^T accumulates via WMMA with per-lane rescale.
// Pipeline per key tile: K loads -> bias loads -> S WMMAs -> V loads ->
// bias add + softmax (VALU hides V latency) -> P build -> O WMMAs.
__global__ __launch_bounds__(128)
void attn_fwd(const bf16_t* __restrict__ Qb, const bf16_t* __restrict__ Kb,
              const bf16_t* __restrict__ VTb, const float* __restrict__ relp,
              const int* __restrict__ mask, bf16_t* __restrict__ Yb) {
  const int lane = threadIdx.x & 31;
  const int wave = threadIdx.x >> 5;
  const int tile = blockIdx.x * 4 + wave;     // NB*NH*(SEQ/16) tiles
  const int qt = tile & 127;
  const int bh = tile >> 7;
  const int h = bh % NH, b = bh / NH;
  const int q0 = qt << 4;
  const int ql = lane & 15;
  const bool hi = lane >= 16;
  const int dof = hi ? 8 : 0;
  const int qg = q0 + ql;

  const bf16_t* Q  = Qb  + ((size_t)(b * NH + h) * SEQ) * DH;
  const bf16_t* K  = Kb  + ((size_t)(b * NH + h) * SK) * DH;
  const bf16_t* VT = VTb + ((size_t)(b * NH + h) * DH) * SK;
  const float*  RP = relp + ((size_t)h * SEQ + qg) * SEQ;
  const int*    MK = mask + ((size_t)b * SEQ + qg) * SEQ;

  // Q^T B-operand: lane = query, K = d. Two chunks for D=64.
  const bf16_t* qrow = Q + (size_t)qg * DH + (hi ? 8 : 0);
  const v16bf qB0 = cat16(ld8(qrow), ld8(qrow + 16));
  const v16bf qB1 = cat16(ld8(qrow + 32), ld8(qrow + 48));

  v8f o[4];
#pragma unroll
  for (int i = 0; i < 4; ++i) o[i] = vzero8();
  float mrun = NEGB, lrun = 0.0f;

  for (int kk = 0; kk < SK; kk += 32) {
    // --- issue K-tile fragment loads first ---
    const bf16_t* k0 = K + (size_t)(kk + ql) * DH + dof;
    const bf16_t* k1 = K + (size_t)(kk + 16 + ql) * DH + dof;
    const v16bf ka00 = cat16(ld8(k0), ld8(k0 + 16));
    const v16bf ka01 = cat16(ld8(k0 + 32), ld8(k0 + 48));
    const v16bf ka10 = cat16(ld8(k1), ld8(k1 + 16));
    const v16bf ka11 = cat16(ld8(k1 + 32), ld8(k1 + 48));

    // --- issue independent bias loads (consumed after S WMMAs) ---
    const bool self = kk < SEQ;
    union { float4 v[4]; float f[16]; } rpu;
    union { int4   v[4]; int   i[16]; } mku;
    if (self) {
      const float* rp = RP + kk + dof;
      const int*   mk = MK + kk + dof;
      rpu.v[0] = *(const float4*)(rp);      rpu.v[1] = *(const float4*)(rp + 4);
      rpu.v[2] = *(const float4*)(rp + 16); rpu.v[3] = *(const float4*)(rp + 20);
      mku.v[0] = *(const int4*)(mk);        mku.v[1] = *(const int4*)(mk + 4);
      mku.v[2] = *(const int4*)(mk + 16);   mku.v[3] = *(const int4*)(mk + 20);
      // stream-prefetch upcoming bias rows (rel_pos/mask are the HBM stream)
      if (kk + 256 < SEQ) {
        __builtin_prefetch(RP + kk + 256, 0, 1);
        __builtin_prefetch(MK + kk + 256, 0, 1);
      }
    }
    if (kk + 32 < SK) __builtin_prefetch(k0 + 32 * DH, 0, 1);

    // --- S^T = K @ Q^T (scale already folded into Q) ---
    v8f s0 = vzero8(), s1 = vzero8();
    s0 = WMMA_BF16(ka00, qB0, s0);
    s0 = WMMA_BF16(ka01, qB1, s0);
    s1 = WMMA_BF16(ka10, qB0, s1);
    s1 = WMMA_BF16(ka11, qB1, s1);

    // --- issue V^T fragment loads; softmax VALU below hides their latency ---
    v16bf va[4];
#pragma unroll
    for (int dt = 0; dt < 4; ++dt) {
      const bf16_t* vr = VT + (size_t)(dt * 16 + ql) * SK + kk + (hi ? 16 : 0);
      va[dt] = cat16(ld8(vr), ld8(vr + 8));
    }

    // --- additive bias: mask -> -BIG, rel_pos on self keys; ref keys 0 ---
    if (self) {
#pragma unroll
      for (int r = 0; r < 8; ++r) {
        s0[r] += mku.i[r]     ? NEGB : rpu.f[r];
        s1[r] += mku.i[8 + r] ? NEGB : rpu.f[8 + r];
      }
    }

    // --- online softmax (per query = per lane; merge paired lane) ---
    float pm = s0[0];
#pragma unroll
    for (int r = 1; r < 8; ++r) pm = fmaxf(pm, s0[r]);
#pragma unroll
    for (int r = 0; r < 8; ++r) pm = fmaxf(pm, s1[r]);
    pm = fmaxf(pm, __shfl_xor(pm, 16, 32));
    const float mn = fmaxf(mrun, pm);
    const float alpha = __expf(mrun - mn);
    v8f p0, p1;
    float ps = 0.0f;
#pragma unroll
    for (int r = 0; r < 8; ++r) {
      p0[r] = __expf(s0[r] - mn); ps += p0[r];
      p1[r] = __expf(s1[r] - mn); ps += p1[r];
    }
    ps += __shfl_xor(ps, 16, 32);
    lrun = lrun * alpha + ps;
    mrun = mn;
#pragma unroll
    for (int i = 0; i < 4; ++i) o[i] = o[i] * alpha;

    // --- build P^T as B-operand (32 keys x 16 queries) ---
    unsigned pk0[4], pk1[4];
#pragma unroll
    for (int i = 0; i < 4; ++i) {
      pk0[i] = pack2bf(p0[2 * i], p0[2 * i + 1]);
      pk1[i] = pack2bf(p1[2 * i], p1[2 * i + 1]);
    }
    union { unsigned u[8]; v16bf v; } pb;
#pragma unroll
    for (int i = 0; i < 4; ++i) {
      const unsigned x0 = (unsigned)__shfl_xor((int)pk0[i], 16, 32);
      const unsigned x1 = (unsigned)__shfl_xor((int)pk1[i], 16, 32);
      pb.u[i]     = hi ? x1 : pk0[i];       // K = kk..kk+15  (lo lanes)
      pb.u[4 + i] = hi ? pk1[i] : x0;       // K = kk+16..+31 (hi lanes)
    }

    // --- O^T += V^T @ P^T ---
#pragma unroll
    for (int dt = 0; dt < 4; ++dt) o[dt] = WMMA_BF16(va[dt], pb.v, o[dt]);
  }

  // --- normalize, write y (B,T,C) bf16 ---
  const float inv = 1.0f / lrun;
  bf16_t* yrow = Yb + ((size_t)(b * SEQ + qg)) * CDIM + h * DH;
#pragma unroll
  for (int dt = 0; dt < 4; ++dt) {
    v8bf wv;
#pragma unroll
    for (int r = 0; r < 8; ++r) wv[r] = (bf16_t)(o[dt][r] * inv);
    *(v8bf*)(yrow + dt * 16 + dof) = wv;
  }
}

// ------------------------------------------------------------------- host ---
extern "C" void kernel_launch(void* const* d_in, const int* in_sizes, int n_in,
                              void* d_out, int out_size, void* d_ws, size_t ws_size,
                              hipStream_t stream) {
  (void)in_sizes; (void)n_in; (void)out_size; (void)ws_size;
  const float* x    = (const float*)d_in[0];
  const int*   mask = (const int*)d_in[1];
  const float* relp = (const float*)d_in[2];
  const float* ref  = (const float*)d_in[3];
  const float* W[6]  = { (const float*)d_in[4],  (const float*)d_in[6],
                         (const float*)d_in[8],  (const float*)d_in[10],
                         (const float*)d_in[12], (const float*)d_in[14] };
  const float* bs[6] = { (const float*)d_in[5],  (const float*)d_in[7],
                         (const float*)d_in[9],  (const float*)d_in[11],
                         (const float*)d_in[13], (const float*)d_in[15] };

  char* p = (char*)d_ws;
  bf16_t* xbf = (bf16_t*)p; p += (size_t)NB * SEQ * CDIM * 2;
  bf16_t* rbf = (bf16_t*)p; p += (size_t)NB * SREF * CDIM * 2;
  bf16_t* wbf[6];
  for (int i = 0; i < 6; ++i) { wbf[i] = (bf16_t*)p; p += (size_t)CDIM * CDIM * 2; }
  bf16_t* qbf = (bf16_t*)p; p += (size_t)NB * NH * SEQ * DH * 2;
  bf16_t* kbf = (bf16_t*)p; p += (size_t)NB * NH * SK * DH * 2;
  bf16_t* vtb = (bf16_t*)p; p += (size_t)NB * NH * DH * SK * 2;
  bf16_t* ybf = (bf16_t*)p; p += (size_t)NB * SEQ * CDIM * 2;

  const int nx = NB * SEQ * CDIM, nr = NB * SREF * CDIM, nw = CDIM * CDIM;
  cvt_f32_bf16<<<(nx + 255) / 256, 256, 0, stream>>>(x, xbf, nx);
  cvt_f32_bf16<<<(nr + 255) / 256, 256, 0, stream>>>(ref, rbf, nr);
  for (int i = 0; i < 6; ++i)
    cvt_f32_bf16<<<(nw + 255) / 256, 256, 0, stream>>>(W[i], wbf[i], nw);

  const float qscale = 0.125f;  // 1/sqrt(D), folded into Q
  // Q : (B,NH,SEQ,DH)
  gemm16x64<<<768, 128, 0, stream>>>(xbf, wbf[0], bs[0], NB * SEQ, CDIM, CDIM,
                                     qscale, 1, 11, SEQ, 0, qbf);
  // K self -> rows [0,SEQ)
  gemm16x64<<<768, 128, 0, stream>>>(xbf, wbf[1], bs[1], NB * SEQ, CDIM, CDIM,
                                     1.0f, 1, 11, SK, 0, kbf);
  // RK -> rows [SEQ,SK)
  gemm16x64<<<192, 128, 0, stream>>>(rbf, wbf[3], bs[3], NB * SREF, CDIM, CDIM,
                                     1.0f, 1, 9, SK, SEQ, kbf);
  // V^T self -> cols [0,SEQ)
  gemm16x64<<<768, 128, 0, stream>>>(xbf, wbf[2], bs[2], NB * SEQ, CDIM, CDIM,
                                     1.0f, 2, 11, SK, 0, vtb);
  // RV^T -> cols [SEQ,SK)
  gemm16x64<<<192, 128, 0, stream>>>(rbf, wbf[4], bs[4], NB * SREF, CDIM, CDIM,
                                     1.0f, 2, 9, SK, SEQ, vtb);
  // attention: NB*NH*(SEQ/16) = 3072 wave-tiles, 4 waves/block
  attn_fwd<<<768, 128, 0, stream>>>(qbf, kbf, vtb, relp, mask, ybf);
  // out = y @ Wo^T + bo  (fp32)
  gemm16x64<<<768, 128, 0, stream>>>(ybf, wbf[5], bs[5], NB * SEQ, CDIM, CDIM,
                                     1.0f, 0, 11, 1, 0, d_out);
}
